// NLP_80831284511084
// MI455X (gfx1250) — matrix operand order repaired and checked by reading, not compiled, not run on hardware
//
#include <hip/hip_runtime.h>

// MI455X / gfx1250, wave32. LSTM LM forward:
//   emb lookup -> LSTM(H=32) over T=512 -> proj to VOCAB=128.
// Kernel 1: serial recurrence, B/16=32 workgroups x 4 waves, WMMA for gates.
// Kernel 2: massively parallel output projection, WMMA f16, writes 134MB out.

typedef float    v8f  __attribute__((ext_vector_type(8)));
typedef float    v4f  __attribute__((ext_vector_type(4)));
typedef float    v2f  __attribute__((ext_vector_type(2)));
typedef _Float16 v16h __attribute__((ext_vector_type(16)));

#define VOCAB 128
#define EMB   8
#define HID   32
#define BB    512
#define TT    512

// ---- fast activations: use CDNA5 v_tanh_f32 trans op when available ----
__device__ __forceinline__ float fast_tanh(float x) {
#if __has_builtin(__builtin_amdgcn_tanhf)
  return __builtin_amdgcn_tanhf(x);
#else
  float e = __expf(2.0f * x);
  return 1.0f - 2.0f / (e + 1.0f);
#endif
}
__device__ __forceinline__ float fast_sigmoid(float x) {
#if __has_builtin(__builtin_amdgcn_tanhf)
  return 0.5f * __builtin_amdgcn_tanhf(0.5f * x) + 0.5f;
#else
  return 1.0f / (1.0f + __expf(-x));
#endif
}

// Pointwise LSTM cell update for accumulator rows {R0, R0+1}; all vector
// element indices are compile-time constants.
template <int R0>
__device__ __forceinline__ void lstm_pointwise(const v8f (&g)[8],
                                               const float (&gbias)[8],
                                               float cst[2][2],
                                               float hv[2][2]) {
#pragma unroll
  for (int j = 0; j < 2; ++j) {
#pragma unroll
    for (int rl = 0; rl < 2; ++rl) {
      const int r = R0 + rl;
      float iv = fast_sigmoid(g[0 + j][r] + gbias[0 + j]);
      float fv = fast_sigmoid(g[2 + j][r] + gbias[2 + j]);
      float gv = fast_tanh   (g[4 + j][r] + gbias[4 + j]);
      float ov = fast_sigmoid(g[6 + j][r] + gbias[6 + j]);
      float cv = fv * cst[j][rl] + iv * gv;
      cst[j][rl] = cv;
      hv[j][rl] = ov * fast_tanh(cv);
    }
  }
}

// Entire 512-step recurrence loop, instantiated once per wave so the row
// indices (R0) stay compile-time constants and cannot be tail-merged back
// into dynamic selects. STORE=true only for wave 0 (publishes h A-frags).
template <int R0, bool STORE>
__device__ __forceinline__ void run_recurrence(
    const int* __restrict__ tokrow,
    const float* __restrict__ lds_emb,
    float (*__restrict__ lds_h)[16 * HID],
    const v16h (&Bhh)[8],
    const v2f (&Bih0)[8], const v2f (&Bih1)[8],
    const float (&gbias)[8],
    _Float16* __restrict__ hout,
    int col, int hi) {
  float cst[2][2] = {{0.f, 0.f}, {0.f, 0.f}};
  v16h a_h = {};            // h_{-1} = 0 A-fragment

  int tok = tokrow[0];      // software-pipelined token fetch
  for (int t = 0; t < TT; ++t) {
    // prefetch next token early; its global-load latency overlaps the
    // WMMA + pointwise work of this step (value unused on last iter).
    const int tok_next = tokrow[(t + 1) & (TT - 1)];

    // embedding A fragments (f32, exact): A[m][k] = emb[tok_m][k]
    const float* erow = lds_emb + tok * EMB + hi * 2;
    v2f ae0, ae1;
    ae0.x = erow[0]; ae0.y = erow[1];
    ae1.x = erow[4]; ae1.y = erow[5];

    // gates = x @ w_ih^T + h @ w_hh^T   (bias added in pointwise phase)
    v8f g[8];
#pragma unroll
    for (int tl = 0; tl < 8; ++tl) {
      v8f c = {};  // inline zero accumulator
      c = __builtin_amdgcn_wmma_f32_16x16x4_f32(false, ae0, false, Bih0[tl],
                                                (short)0, c, false, false);
      c = __builtin_amdgcn_wmma_f32_16x16x4_f32(false, ae1, false, Bih1[tl],
                                                (short)0, c, false, false);
      c = __builtin_amdgcn_wmma_f32_16x16x32_f16(false, a_h, false, Bhh[tl],
                                                 (short)0, c, false, false);
      g[tl] = c;
    }

    // nonlinearity on this wave's static row subset
    float hv[2][2];
    lstm_pointwise<R0>(g, gbias, cst, hv);

    // publish h rows to ping-pong LDS buffer
    float* buf = lds_h[t & 1];
#pragma unroll
    for (int j = 0; j < 2; ++j)
#pragma unroll
      for (int rl = 0; rl < 2; ++rl)
        buf[(R0 + rl + 8 * hi) * HID + j * 16 + col] = hv[j][rl];
    __syncthreads();

    // rebuild h_t as f16 A-fragment (needed for next step's WMMA)
    {
      const v4f* rowv = (const v4f*)(buf + col * HID + hi * 16);
      v4f x0 = rowv[0];  // K ks..ks+3
      v4f x1 = rowv[1];  // K ks+4..ks+7
      v4f x2 = rowv[2];  // K ks+8..ks+11
      v4f x3 = rowv[3];  // K ks+12..ks+15
      v16h a;
#pragma unroll
      for (int j = 0; j < 4; ++j) {
        a[j]      = (_Float16)x0[j];
        a[4 + j]  = (_Float16)x1[j];
        a[8 + j]  = (_Float16)x2[j];
        a[12 + j] = (_Float16)x3[j];
      }
      a_h = a;
    }

    if (STORE) *(v16h*)(hout + (size_t)t * 512) = a_h;
    tok = tok_next;
  }
}

// =====================================================================
// Kernel 1: LSTM recurrence.
// Each block owns 16 batch rows. 4 waves duplicate the WMMAs (cheap, XDL
// co-executes with VALU) but split the gate nonlinearity by C-VGPR row
// (wave w handles accumulator rows 2w, 2w+1) to shorten the serial VALU
// chain ~4x. Wave id is readfirstlane'd -> scalar branches, and each wave
// runs its own full-loop template instantiation.
//
// Gate layout: gates[m][n], n in [0,128): tiles tl=0..7 of 16 cols.
//   i = tiles 0-1, f = 2-3, g = 4-5, o = 6-7.
// C/D 16x16 f32 tile: VGPR r, lane l -> (M = r + 8*(l>>4), N = l&15).
// f16 A 16x32: lane l -> row M = l&15; halves K[ks..ks+15],
//   ks = 16*(l>>4).
// f16 B 32x16: lane l -> col N = l&15; halves K[16*(l>>4) .. +15].
// f32 A 16x4 : VGPR v, lane l -> (M = l&15, K = v + 2*(l>>4)).
// f32 B 4x16 : VGPR v, lane l -> (K = v + 2*(l>>4), N = l&15).
// =====================================================================
__global__ __launch_bounds__(128)
void lstm_recurrence_kernel(const int*   __restrict__ tokens,
                            const float* __restrict__ emb,
                            const float* __restrict__ w_ih,
                            const float* __restrict__ w_hh,
                            const float* __restrict__ b_ih,
                            const float* __restrict__ b_hh,
                            _Float16*    __restrict__ hbuf) {
  __shared__ __align__(16) float lds_emb[VOCAB * EMB];   // 4 KB
  __shared__ __align__(16) float lds_h[2][16 * HID];     // ping-pong 2x2 KB

  const int tid   = threadIdx.x;
  const int lane  = tid & 31;
  const int wave  = __builtin_amdgcn_readfirstlane(tid >> 5);  // uniform 0..3
  const int col   = lane & 15;         // N (or M for A-frags)
  const int hi    = lane >> 4;         // lane half select
  const int btile = blockIdx.x;        // 0..31
  const int b0    = btile * 16;

  // ---- stage embedding table into LDS (vectorized) ----
  for (int i = tid; i < (VOCAB * EMB) / 4; i += 128)
    ((v4f*)lds_emb)[i] = ((const v4f*)emb)[i];

  // ---- per-lane gate bias: n = tl*16 + col ----
  float gbias[8];
#pragma unroll
  for (int tl = 0; tl < 8; ++tl) {
    int n = tl * 16 + col;
    gbias[tl] = b_ih[n] + b_hh[n];
  }

  // ---- B fragments for h @ w_hh^T : B[k][n] = w_hh[n][k], f16 ----
  v16h Bhh[8];
#pragma unroll
  for (int tl = 0; tl < 8; ++tl) {
    const float* src = w_hh + (tl * 16 + col) * HID + hi * 16;
    v16h b;
#pragma unroll
    for (int j = 0; j < 16; ++j) b[j] = (_Float16)src[j];
    Bhh[tl] = b;
  }

  // ---- B fragments for x @ w_ih^T : B[k][n] = w_ih[n][k], f32, K=4 x2 ----
  v2f Bih0[8], Bih1[8];
#pragma unroll
  for (int tl = 0; tl < 8; ++tl) {
    const float* src = w_ih + (tl * 16 + col) * EMB;
    v2f x0, x1;
    x0.x = src[hi * 2 + 0];     x0.y = src[hi * 2 + 1];      // K 0..3
    x1.x = src[4 + hi * 2 + 0]; x1.y = src[4 + hi * 2 + 1];  // K 4..7
    Bih0[tl] = x0;
    Bih1[tl] = x1;
  }

  __syncthreads();

  const int* tokrow = tokens + (b0 + col) * TT;
  _Float16* hout = hbuf + (size_t)btile * TT * 512 + lane * 16;  // 512 h/step

  // one full-loop instantiation per wave; scalar (uniform) dispatch
  if (wave == 0)
    run_recurrence<0, true >(tokrow, lds_emb, lds_h, Bhh, Bih0, Bih1, gbias,
                             hout, col, hi);
  else if (wave == 1)
    run_recurrence<2, false>(tokrow, lds_emb, lds_h, Bhh, Bih0, Bih1, gbias,
                             hout, col, hi);
  else if (wave == 2)
    run_recurrence<4, false>(tokrow, lds_emb, lds_h, Bhh, Bih0, Bih1, gbias,
                             hout, col, hi);
  else
    run_recurrence<6, false>(tokrow, lds_emb, lds_h, Bhh, Bih0, Bih1, gbias,
                             hout, col, hi);
}

// =====================================================================
// Kernel 2: output projection  out[b,t,:] = h[b,t,:] @ w_out^T + b_out.
// 2048 independent waves; each owns (btile, 8 consecutive t). A-fragments
// come pre-packed from kernel 1 (2x b128 loads/lane), 8 f16 WMMAs per t,
// coalesced 64B store runs -> saturates HBM on the 134 MB output.
// =====================================================================
__global__ __launch_bounds__(256)
void lstm_output_kernel(const _Float16* __restrict__ hbuf,
                        const float*    __restrict__ w_out,
                        const float*    __restrict__ b_out,
                        float*          __restrict__ out) {
  const int tid  = threadIdx.x;
  const int lane = tid & 31;
  const int wave = tid >> 5;
  const int col  = lane & 15;
  const int hi   = lane >> 4;

  const int job   = blockIdx.x * 8 + wave;  // 0..2047
  const int btile = job >> 6;               // 0..31
  const int t0    = (job & 63) * 8;

  // B fragments for w_out^T : B[k][v] = w_out[v][k], f16
  v16h Bo[8];
  float obias[8];
#pragma unroll
  for (int tl = 0; tl < 8; ++tl) {
    int v = tl * 16 + col;
    obias[tl] = b_out[v];
    const float* src = w_out + v * HID + hi * 16;
    v16h b;
#pragma unroll
    for (int j = 0; j < 16; ++j) b[j] = (_Float16)src[j];
    Bo[tl] = b;
  }

  const _Float16* hsrc = hbuf + ((size_t)btile * TT + t0) * 512 + lane * 16;

  for (int i = 0; i < 8; ++i) {
    const int t = t0 + i;
    v16h a = *(const v16h*)(hsrc + (size_t)i * 512);
#pragma unroll
    for (int tl = 0; tl < 8; ++tl) {
      v8f c = {};
      c = __builtin_amdgcn_wmma_f32_16x16x32_f16(false, a, false, Bo[tl],
                                                 (short)0, c, false, false);
#pragma unroll
      for (int r = 0; r < 8; ++r) {
        int brow = btile * 16 + r + 8 * hi;
        out[((size_t)brow * TT + t) * VOCAB + tl * 16 + col] =
            c[r] + obias[tl];
      }
    }
  }
}

extern "C" void kernel_launch(void* const* d_in, const int* in_sizes, int n_in,
                              void* d_out, int out_size, void* d_ws, size_t ws_size,
                              hipStream_t stream) {
  const int*   tokens = (const int*)  d_in[0];
  const float* emb    = (const float*)d_in[1];
  const float* w_ih   = (const float*)d_in[2];
  const float* w_hh   = (const float*)d_in[3];
  const float* b_ih   = (const float*)d_in[4];
  const float* b_hh   = (const float*)d_in[5];
  const float* w_out  = (const float*)d_in[6];
  const float* b_out  = (const float*)d_in[7];
  float* out = (float*)d_out;

  // workspace: h in f16 A-fragment layout, (B/16)*T*1024 bytes = 16.78 MB
  _Float16* hbuf = (_Float16*)d_ws;

  lstm_recurrence_kernel<<<BB / 16, 128, 0, stream>>>(tokens, emb, w_ih, w_hh,
                                                      b_ih, b_hh, hbuf);
  lstm_output_kernel<<<(BB / 16) * (TT / 8) / 8, 256, 0, stream>>>(hbuf, w_out,
                                                                   b_out, out);
}
